// Attention_65249143161114
// MI455X (gfx1250) — matrix operand order repaired and checked by reading, not compiled, not run on hardware
//
#include <hip/hip_runtime.h>
#include <hip/hip_bf16.h>
#include <stdint.h>

// ---------------------------------------------------------------------------
// Causal MHA forward for MI455X (gfx1250, wave32, WMMA 16x16x32 f16,
// async global->LDS staging with double buffering).
// B=4, S=2048, H=16, DH=64, DM=1024.
// ---------------------------------------------------------------------------

typedef __attribute__((ext_vector_type(16))) _Float16 v16h;
typedef __attribute__((ext_vector_type(8)))  float    v8f;

#define WMMA_F16(A, B, C) \
    __builtin_amdgcn_wmma_f32_16x16x32_f16(false, (A), false, (B), (short)0, (C), false, false)

constexpr int Bc = 4;
constexpr int Sc = 2048;
constexpr int Hc = 16;
constexpr int DHc = 64;
constexpr int DMc = 1024;
constexpr int ROWS = Bc * Sc;          // 8192

// ---------------------------------------------------------------------------
// CDNA5 async copy: global -> LDS, tracked by ASYNCcnt (ISA ch.10 / §15.18.3).
// lds_addr: LDS byte offset (low 32 bits of generic pointer, per aperture rule)
// ---------------------------------------------------------------------------
__device__ __forceinline__ void async_load_b128(uint32_t lds_addr, const void* gaddr) {
    asm volatile("global_load_async_to_lds_b128 %0, %1, off"
                 :: "v"(lds_addr), "v"(gaddr)
                 : "memory");
}
__device__ __forceinline__ void wait_async0() {
    asm volatile("s_wait_asynccnt 0x0" ::: "memory");
}
__device__ __forceinline__ uint32_t lds_off(const void* p) {
    return (uint32_t)(uintptr_t)p;     // LDS generic addr keeps offset in [31:0]
}

// ---------------------------------------------------------------------------
// Stage 1: converts
// ---------------------------------------------------------------------------
__global__ void cvt_f32_to_f16(const float* __restrict__ src,
                               _Float16* __restrict__ dst, int n) {
    int i = blockIdx.x * blockDim.x + threadIdx.x;
    if (i < n) dst[i] = (_Float16)src[i];
}

// Q/K/V weights: (H, M, D) -> Wt[m][h*64+d], f16
__global__ void cvt_wqkv(const float* __restrict__ wq,
                         const float* __restrict__ wk,
                         const float* __restrict__ wv,
                         _Float16* __restrict__ oq,
                         _Float16* __restrict__ ok,
                         _Float16* __restrict__ ov) {
    int i = blockIdx.x * blockDim.x + threadIdx.x;   // over 1024*1024
    int m = i >> 10;
    int c = i & 1023;
    int h = c >> 6;
    int d = c & 63;
    size_t si = ((size_t)h * DMc + m) * DHc + d;
    const float* w = (blockIdx.y == 0) ? wq : (blockIdx.y == 1) ? wk : wv;
    _Float16*    o = (blockIdx.y == 0) ? oq : (blockIdx.y == 1) ? ok : ov;
    o[i] = (_Float16)w[si];
}

// ---------------------------------------------------------------------------
// Fragment helpers (layouts per CDNA5 ISA 7.12.2, wave32)
//   A (16x32 f16): lane m = lane&15, half = lane>>4,
//                  a[2r],a[2r+1] = A[m][k], k = 2*(r&3) + 8*half + 16*(r>>2)
//   B (32x16 f16): n = lane&15, b[e] = B[k][n], k = (e&7) + 8*half + 16*(e>>3)
//   C (16x16 f32): c[r] = C[r + 8*half][lane&15]
// ---------------------------------------------------------------------------
__device__ __forceinline__ v16h load_A_frag(const _Float16* __restrict__ src, int ld) {
    int lane = threadIdx.x & 31;
    int m = lane & 15, hh = lane >> 4;
    v16h a;
#pragma unroll
    for (int r = 0; r < 8; ++r) {
        int k = ((r & 3) << 1) + (hh << 3) + ((r >> 2) << 4);
        a[2 * r]     = src[m * ld + k];
        a[2 * r + 1] = src[m * ld + k + 1];
    }
    return a;
}

// ---------------------------------------------------------------------------
// Stage 2 & 4: GEMM  C[8192,1024] = A(f16) @ W(f16) + bias(f32)
// block = 256 threads (8 waves), tile 128x128, K-step 32, async double buffer.
// wave grid 4(M) x 2(N): each wave -> 2 M16-subtiles x 4 N16-subtiles.
// ---------------------------------------------------------------------------
template <typename OutT>
__device__ __forceinline__ void gemm_body(const _Float16* __restrict__ A,
                                          const _Float16* __restrict__ W,
                                          const float* __restrict__ bias,
                                          OutT* __restrict__ Out) {
    constexpr int LDA = 40;    // 128x32 tile rows, 80B stride (16B-aligned)
    constexpr int LDW = 136;   // 32x128 tile rows, 272B stride (16B-aligned)
    __shared__ _Float16 As[2][128 * LDA];
    __shared__ _Float16 Ws[2][32 * LDW];

    const int t = threadIdx.x;
    const int lane = t & 31;
    const int wave = t >> 5;
    const int wm = wave & 3;                // M subtile group (32 rows)
    const int wn = wave >> 2;               // N subtile group (64 cols)
    const int mBase = blockIdx.y * 128;
    const int nBase = blockIdx.x * 128;

    // issue one stage = 4 async b128 ops per wave (per-lane 16B chunks)
    auto stage = [&](int kb, int buf) {
#pragma unroll
        for (int it = 0; it < 2; ++it) {
            int i = t + it * 256;               // A tile: 128 rows x 4 chunks
            int row = i >> 2, c = i & 3;
            async_load_b128(lds_off(&As[buf][row * LDA + c * 8]),
                            &A[(size_t)(mBase + row) * DMc + kb + c * 8]);
        }
#pragma unroll
        for (int it = 0; it < 2; ++it) {
            int i = t + it * 256;               // W tile: 32 rows x 16 chunks
            int row = i >> 4, c = i & 15;
            async_load_b128(lds_off(&Ws[buf][row * LDW + c * 8]),
                            &W[(size_t)(kb + row) * DMc + nBase + c * 8]);
        }
    };

    v8f acc[2][4] = {};

    constexpr int NIT = DMc / 32;   // 32 k-steps
    stage(0, 0);
    for (int it = 0; it < NIT; ++it) {
        wait_async0();              // this wave's current tile landed in LDS
        __syncthreads();            // everyone's tile landed; prev reads done
        if (it + 1 < NIT) stage((it + 1) * 32, (it + 1) & 1);   // overlap

        const _Float16* AsB = As[it & 1];
        const _Float16* WsB = Ws[it & 1];

        v16h bf[4];
#pragma unroll
        for (int j = 0; j < 4; ++j) {
            const _Float16* src = &WsB[wn * 64 + j * 16 + (lane & 15)];
#pragma unroll
            for (int e = 0; e < 16; ++e) bf[j][e] = src[((e & 7) + ((lane >> 4) << 3) + ((e >> 3) << 4)) * LDW];
        }
#pragma unroll
        for (int i = 0; i < 2; ++i) {
            v16h af = load_A_frag(&AsB[(wm * 32 + i * 16) * LDA], LDA);
#pragma unroll
            for (int j = 0; j < 4; ++j) acc[i][j] = WMMA_F16(af, bf[j], acc[i][j]);
        }
    }

    // ---- epilogue: bias add + store ----
    const int mrow0 = mBase + wm * 32;
    const int ncol0 = nBase + wn * 64;
    const int hh = lane >> 4, nn = lane & 15;
#pragma unroll
    for (int i = 0; i < 2; ++i)
#pragma unroll
        for (int j = 0; j < 4; ++j) {
            int col = ncol0 + j * 16 + nn;
            float bv = bias[col];
#pragma unroll
            for (int r = 0; r < 8; ++r) {
                int row = mrow0 + i * 16 + r + 8 * hh;
                Out[(size_t)row * DMc + col] = (OutT)(acc[i][j][r] + bv);
            }
        }
}

__global__ __launch_bounds__(256)
void gemm_qkv_kernel(const _Float16* __restrict__ X,
                     const _Float16* __restrict__ Wq, const _Float16* __restrict__ Wk,
                     const _Float16* __restrict__ Wv,
                     const float* __restrict__ bq, const float* __restrict__ bk,
                     const float* __restrict__ bv,
                     _Float16* __restrict__ Oq, _Float16* __restrict__ Ok,
                     _Float16* __restrict__ Ov) {
    const int z = blockIdx.z;
    const _Float16* W   = (z == 0) ? Wq : (z == 1) ? Wk : Wv;
    const float*    bia = (z == 0) ? bq : (z == 1) ? bk : bv;
    _Float16*       O   = (z == 0) ? Oq : (z == 1) ? Ok : Ov;
    gemm_body<_Float16>(X, W, bia, O);
}

__global__ __launch_bounds__(256)
void gemm_out_kernel(const _Float16* __restrict__ Attn,
                     const _Float16* __restrict__ Wo,
                     const float* __restrict__ ob,
                     float* __restrict__ Out) {
    gemm_body<float>(Attn, Wo, ob, Out);
}

// ---------------------------------------------------------------------------
// Stage 3: flash attention, causal, per (b, h, 128 q-rows)
// block = 256 threads (8 waves); each wave owns 16 q rows.
// K/V 32-key tiles async-staged to double-buffered LDS; online softmax;
// P routed C-layout -> A-layout through wave-private LDS (in-order per wave).
// ---------------------------------------------------------------------------
__global__ __launch_bounds__(256)
void flash_attn_kernel(const _Float16* __restrict__ Q,
                       const _Float16* __restrict__ K,
                       const _Float16* __restrict__ V,
                       _Float16* __restrict__ Oattn) {
    constexpr int LDK = 72;   // 144B row stride (16B-aligned for async b128)
    constexpr int LDP = 34;
    __shared__ _Float16 KV[2][2][32 * LDK];   // [buf][K/V][tile]
    __shared__ _Float16 Pl[8][16 * LDP];

    const int bI    = blockIdx.z;
    const int h     = blockIdx.y;
    const int qBase = blockIdx.x * 128;
    const int t = threadIdx.x, lane = t & 31, wave = t >> 5;
    const int qRow0 = qBase + wave * 16;
    const size_t base = (size_t)bI * Sc * DMc + h * DHc;   // row-major [8192][1024]

    // issue one K/V stage = 2 async b128 ops per wave (per-lane 16B chunks)
    auto stage = [&](int kb, int buf) {
        int row = t >> 3, c = t & 7;          // 32 rows x 8 chunks
        const size_t g = base + (size_t)(kb + row) * DMc + c * 8;
        async_load_b128(lds_off(&KV[buf][0][row * LDK + c * 8]), K + g);
        async_load_b128(lds_off(&KV[buf][1][row * LDK + c * 8]), V + g);
    };

    // Q fragments for d-chunks 0 and 32 (kept in registers for whole kernel)
    v16h qa[2];
    {
        const _Float16* qs = Q + base + (size_t)qRow0 * DMc;
        int m = lane & 15, hh = lane >> 4;
#pragma unroll
        for (int i = 0; i < 2; ++i)
#pragma unroll
            for (int r = 0; r < 8; ++r) {
                int k = i * 32 + ((r & 3) << 1) + (hh << 3) + ((r >> 2) << 4);
                qa[i][2 * r]     = qs[m * DMc + k];
                qa[i][2 * r + 1] = qs[m * DMc + k + 1];
            }
    }

    float mrun[8], lrun[8];
    v8f acc[4] = {};
#pragma unroll
    for (int r = 0; r < 8; ++r) { mrun[r] = -3.0e38f; lrun[r] = 0.0f; }

    const int nIter = (qBase + 128) >> 5;     // block-uniform causal bound
    stage(0, 0);
    for (int it = 0; it < nIter; ++it) {
        wait_async0();
        __syncthreads();
        if (it + 1 < nIter) stage((it + 1) * 32, (it + 1) & 1);

        const int kb = it * 32;
        const _Float16* Kt = KV[it & 1][0];
        const _Float16* Vt = KV[it & 1][1];

        // ---- scores: S = (Q @ K^T) / 8, causal mask to -1e5 ----
        float sc[2][8];
#pragma unroll
        for (int j = 0; j < 2; ++j) {
            v8f s = {};
#pragma unroll
            for (int dc = 0; dc < 2; ++dc) {
                // B[kdim=d][n=key]: b[e] = Kt[16j+(lane&15)][32*dc + (e&7) + 8*(lane>>4) + 16*(e>>3)]
                v16h bfrag;
                const _Float16* src = &Kt[(j * 16 + (lane & 15)) * LDK + dc * 32];
#pragma unroll
                for (int e = 0; e < 16; ++e) bfrag[e] = src[(e & 7) + ((lane >> 4) << 3) + ((e >> 3) << 4)];
                s = WMMA_F16(qa[dc], bfrag, s);
            }
            int col = kb + j * 16 + (lane & 15);
#pragma unroll
            for (int r = 0; r < 8; ++r) {
                int qrow = qRow0 + r + 8 * (lane >> 4);
                float v = s[r] * 0.125f;               // 1/sqrt(64)
                sc[j][r] = (col > qrow) ? -100000.0f : v;
            }
        }

        // ---- online softmax (row reductions across 16-lane halves) ----
        float pv[2][8];
#pragma unroll
        for (int r = 0; r < 8; ++r) {
            float rowm = fmaxf(sc[0][r], sc[1][r]);
#pragma unroll
            for (int off = 8; off >= 1; off >>= 1)
                rowm = fmaxf(rowm, __shfl_xor(rowm, off, 32));
            float mnew  = fmaxf(mrun[r], rowm);
            float alpha = __expf(mrun[r] - mnew);
            mrun[r] = mnew;
            float p0 = __expf(sc[0][r] - mnew);
            float p1 = __expf(sc[1][r] - mnew);
            pv[0][r] = p0; pv[1][r] = p1;
            float rs = p0 + p1;
#pragma unroll
            for (int off = 8; off >= 1; off >>= 1)
                rs += __shfl_xor(rs, off, 32);
            lrun[r] = lrun[r] * alpha + rs;
#pragma unroll
            for (int n = 0; n < 4; ++n) acc[n][r] *= alpha;
        }

        // ---- P: C-layout -> A-layout via wave-private LDS ----
        {
            int nn = lane & 15, hh = lane >> 4;
            _Float16* pw = &Pl[wave][0];
#pragma unroll
            for (int j = 0; j < 2; ++j)
#pragma unroll
                for (int r = 0; r < 8; ++r)
                    pw[(r + 8 * hh) * LDP + j * 16 + nn] = (_Float16)pv[j][r];
        }
        v16h pa = load_A_frag(&Pl[wave][0], LDP);

        // ---- acc += P @ V ----
#pragma unroll
        for (int n = 0; n < 4; ++n) {
            v16h bfrag;
            const _Float16* src = &Vt[n * 16 + (lane & 15)];
#pragma unroll
            for (int e = 0; e < 16; ++e) bfrag[e] = src[((e & 7) + ((lane >> 4) << 3) + ((e >> 3) << 4)) * LDK];
            acc[n] = WMMA_F16(pa, bfrag, acc[n]);
        }
    }

    // ---- epilogue: normalize and store f16 ----
    {
        int nn = lane & 15, hh = lane >> 4;
#pragma unroll
        for (int r = 0; r < 8; ++r) {
            float inv = 1.0f / lrun[r];
            int row = qRow0 + r + 8 * hh;
            _Float16* dst = Oattn + base + (size_t)row * DMc;
#pragma unroll
            for (int n = 0; n < 4; ++n)
                dst[n * 16 + nn] = (_Float16)(acc[n][r] * inv);
        }
    }
}

// ---------------------------------------------------------------------------
// Host launcher
// ---------------------------------------------------------------------------
extern "C" void kernel_launch(void* const* d_in, const int* in_sizes, int n_in,
                              void* d_out, int out_size, void* d_ws, size_t ws_size,
                              hipStream_t stream) {
    const float* x   = (const float*)d_in[0];
    const float* Qw  = (const float*)d_in[1];
    const float* Qb  = (const float*)d_in[2];
    const float* Kw  = (const float*)d_in[3];
    const float* Kb  = (const float*)d_in[4];
    const float* Vw  = (const float*)d_in[5];
    const float* Vb  = (const float*)d_in[6];
    const float* Ow  = (const float*)d_in[7];
    const float* Ob  = (const float*)d_in[8];
    float* out = (float*)d_out;

    // workspace carve-up (256B aligned)
    char* p = (char*)d_ws;
    size_t off = 0;
    auto carve = [&](size_t bytes) -> void* {
        void* r = p + off;
        off += (bytes + 255) & ~(size_t)255;
        return r;
    };
    _Float16* xh  = (_Float16*)carve((size_t)ROWS * DMc * 2);  // 16 MB
    _Float16* wqh = (_Float16*)carve((size_t)DMc * DMc * 2);   //  2 MB
    _Float16* wkh = (_Float16*)carve((size_t)DMc * DMc * 2);
    _Float16* wvh = (_Float16*)carve((size_t)DMc * DMc * 2);
    _Float16* woh = (_Float16*)carve((size_t)DMc * DMc * 2);
    _Float16* qh  = (_Float16*)carve((size_t)ROWS * DMc * 2);  // 16 MB
    _Float16* kh  = (_Float16*)carve((size_t)ROWS * DMc * 2);
    _Float16* vh  = (_Float16*)carve((size_t)ROWS * DMc * 2);
    _Float16* ah  = (_Float16*)carve((size_t)ROWS * DMc * 2);
    (void)ws_size; (void)n_in; (void)in_sizes; (void)out_size;

    // 1) converts
    cvt_f32_to_f16<<<(ROWS * DMc) / 256, 256, 0, stream>>>(x, xh, ROWS * DMc);
    cvt_f32_to_f16<<<(DMc * DMc) / 256, 256, 0, stream>>>(Ow, woh, DMc * DMc);
    cvt_wqkv<<<dim3((DMc * DMc) / 256, 3), 256, 0, stream>>>(Qw, Kw, Vw, wqh, wkh, wvh);

    // 2) QKV projections (grid: N-tiles=8, M-tiles=64, z selects Q/K/V)
    gemm_qkv_kernel<<<dim3(DMc / 128, ROWS / 128, 3), 256, 0, stream>>>(
        xh, wqh, wkh, wvh, Qb, Kb, Vb, qh, kh, vh);

    // 3) causal flash attention (grid: S/128 q-tiles, H heads, B batches)
    flash_attn_kernel<<<dim3(Sc / 128, Hc, Bc), 256, 0, stream>>>(qh, kh, vh, ah);

    // 4) output projection
    gemm_out_kernel<<<dim3(DMc / 128, ROWS / 128), 256, 0, stream>>>(ah, woh, Ob, out);
}
